// NestedAttention_23218593202423
// MI455X (gfx1250) — matrix operand order repaired and checked by reading, not compiled, not run on hardware
//
#include <hip/hip_runtime.h>

// ---------------------------------------------------------------------------
// NestedAttention for MI455X (gfx1250, wave32, WMMA bf16).
// All matmuls (QKV GEMMs, QK^T, P*V, output GEMM) run through
// v_wmma_f32_16x16x32_bf16 with fp32 accumulation. Softmax / RMSNorm / RoPE
// stay in fp32 VALU. Attention uses 32-query-row tiles per wave so every
// loaded K/V fragment feeds two WMMAs. Requires >= 136 MB workspace.
// ---------------------------------------------------------------------------

typedef __bf16 bf16;
typedef __attribute__((ext_vector_type(16))) __bf16 v16bf;
typedef __attribute__((ext_vector_type(8)))  __bf16 v8bf;
typedef __attribute__((ext_vector_type(8)))  float  v8f;

#define B_    2
#define NS    2048
#define DIM_  1024
#define H_    16
#define DH_   64
#define MROWS (B_ * NS)                 // 4096
#define EPSF  1.1920928955078125e-07f
#define NEG_BIG (-1.0e30f)

static __device__ __forceinline__ bf16 f2bf(float x) { return (bf16)x; }

static __device__ __forceinline__ v8f vzero8() {
  v8f z;
#pragma unroll
  for (int i = 0; i < 8; ++i) z[i] = 0.f;
  return z;
}

static __device__ __forceinline__ v16bf cat8(v8bf lo, v8bf hi) {
  return __builtin_shufflevector(lo, hi, 0, 1, 2, 3, 4, 5, 6, 7,
                                 8, 9, 10, 11, 12, 13, 14, 15);
}

static __device__ __forceinline__ void cstore(bf16* p, float v) { *p = f2bf(v); }
static __device__ __forceinline__ void cstore(float* p, float v) { *p = v; }

// ---------------------------------------------------------------------------
// 1) RMSNorm over rows of DIM_ fp32 -> bf16
// ---------------------------------------------------------------------------
__global__ __launch_bounds__(256) void rmsnorm_rows_kernel(
    const float* __restrict__ in, const float* __restrict__ w,
    bf16* __restrict__ out, int cols) {
  const int row = blockIdx.x;
  const float* x = in + (size_t)row * cols;
  float ss = 0.f;
  for (int c = threadIdx.x; c < cols; c += blockDim.x) {
    float v = x[c];
    ss += v * v;
  }
#pragma unroll
  for (int m = 16; m >= 1; m >>= 1) ss += __shfl_xor(ss, m, 32);
  __shared__ float wsum[8];
  const int wid = threadIdx.x >> 5;
  if ((threadIdx.x & 31) == 0) wsum[wid] = ss;
  __syncthreads();
  float tot = 0.f;
  const int nw = blockDim.x >> 5;
  for (int i = 0; i < nw; ++i) tot += wsum[i];
  const float r = rsqrtf(tot / (float)cols + EPSF);
  bf16* o = out + (size_t)row * cols;
  for (int c = threadIdx.x; c < cols; c += blockDim.x)
    o[c] = f2bf(x[c] * r * w[c]);
}

// ---------------------------------------------------------------------------
// 2) Weight transpose + cast: W[K][Nout] fp32 -> WT[Nout][K] bf16
// ---------------------------------------------------------------------------
__global__ __launch_bounds__(256) void transpose_cast_kernel(
    const float* __restrict__ W, bf16* __restrict__ WT, int K, int Nout) {
  const size_t total = (size_t)K * (size_t)Nout;
  for (size_t t = (size_t)blockIdx.x * blockDim.x + threadIdx.x; t < total;
       t += (size_t)gridDim.x * blockDim.x) {
    const size_t n = t / (size_t)K, k = t % (size_t)K;
    WT[t] = f2bf(W[k * (size_t)Nout + n]);
  }
}

// ---------------------------------------------------------------------------
// 3) bf16 WMMA GEMM: C[M][Nout] = A[M][K] * BT[Nout][K]^T
//    8 waves / block; each wave computes a 32x64 tile (2x4 accumulators).
// ---------------------------------------------------------------------------
template <typename OutT>
__global__ __launch_bounds__(256) void gemm_bf16_kernel(
    const bf16* __restrict__ A, const bf16* __restrict__ BT,
    OutT* __restrict__ C, int M, int K, int Nout) {
  const int lane = threadIdx.x & 31;
  const int w    = threadIdx.x >> 5;
  const int half = lane >> 4;
  const int lr   = lane & 15;
  const int wr = w >> 2, wc = w & 3;
  const int m0 = blockIdx.y * 64 + wr * 32;
  const int n0 = blockIdx.x * 256 + wc * 64;

  v8f acc[2][4];
#pragma unroll
  for (int i = 0; i < 2; ++i)
#pragma unroll
    for (int j = 0; j < 4; ++j) acc[i][j] = vzero8();

  for (int k0 = 0; k0 < K; k0 += 32) {
    // prefetch next K tile into the cache hierarchy (global_prefetch_b8)
    if (k0 + 32 < K) {
      __builtin_prefetch(A + (size_t)(m0 + lane) * K + k0 + 32, 0, 1);
      __builtin_prefetch(BT + (size_t)(n0 + lane) * K + k0 + 32, 0, 1);
      __builtin_prefetch(BT + (size_t)(n0 + 32 + lane) * K + k0 + 32, 0, 1);
    }
    v16bf af[2];
#pragma unroll
    for (int ms = 0; ms < 2; ++ms) {
      const bf16* p = A + (size_t)(m0 + ms * 16 + lr) * K + k0 + 16 * half;
      v8bf lo = *(const v8bf*)p;
      v8bf hi = *(const v8bf*)(p + 8);
      af[ms] = cat8(lo, hi);
    }
    v16bf bfg[4];
#pragma unroll
    for (int ns = 0; ns < 4; ++ns) {
      const bf16* p = BT + (size_t)(n0 + ns * 16 + lr) * K + k0 + 16 * half;
      bfg[ns] = *(const v16bf*)p;
    }
#pragma unroll
    for (int ms = 0; ms < 2; ++ms)
#pragma unroll
      for (int ns = 0; ns < 4; ++ns)
        acc[ms][ns] = __builtin_amdgcn_wmma_f32_16x16x32_bf16(
            false, af[ms], false, bfg[ns], (short)0, acc[ms][ns], false, false);
  }

#pragma unroll
  for (int ms = 0; ms < 2; ++ms)
#pragma unroll
    for (int ns = 0; ns < 4; ++ns)
#pragma unroll
      for (int r = 0; r < 8; ++r) {
        const int row = m0 + ms * 16 + r + 8 * half;
        const int col = n0 + ns * 16 + lr;
        cstore(&C[(size_t)row * Nout + col], acc[ms][ns][r]);
      }
}

// ---------------------------------------------------------------------------
// 4) RoPE + re-layout: strided bf16 in -> [b][h][n][64] bf16 out. 32 thr/row.
// ---------------------------------------------------------------------------
__global__ __launch_bounds__(32) void rope_cast_kernel(
    const bf16* __restrict__ in, bf16* __restrict__ out,
    long long sB, long long sH, long long sN) {
  const int gid = blockIdx.x;
  const int n = gid % NS;
  const int t = gid / NS;
  const int h = t % H_;
  const int b = t / H_;
  const int i = threadIdx.x;  // rotary pair index 0..31
  const bf16* p = in + (size_t)b * sB + (size_t)h * sH + (size_t)n * sN;
  float x0 = (float)p[2 * i], x1 = (float)p[2 * i + 1];
  const float inv = __powf(10000.f, -(float)i / 32.f);
  const float ang = (float)n * inv;
  const float c = __cosf(ang), s = __sinf(ang);
  bf16* o = out + ((size_t)(b * H_ + h) * NS + n) * DH_;
  o[2 * i]     = f2bf(x0 * c - x1 * s);
  o[2 * i + 1] = f2bf(x1 * c + x0 * s);
}

// ---------------------------------------------------------------------------
// 5) Per-head RMSNorm(64) + RoPE: strided bf16 in -> [(br*B+b)][h][n][64]
// ---------------------------------------------------------------------------
__global__ __launch_bounds__(32) void knorm_rope_kernel(
    const bf16* __restrict__ in, const float* __restrict__ w,
    bf16* __restrict__ out, long long sBr, long long sB, long long sH,
    long long sN, int wStride) {
  const int gid = blockIdx.x;
  const int n = gid % NS;
  int t = gid / NS;
  const int h = t % H_;
  t /= H_;
  const int b = t % B_;
  const int br = t / B_;
  const int i = threadIdx.x;
  const bf16* p = in + (size_t)br * sBr + (size_t)b * sB + (size_t)h * sH +
                  (size_t)n * sN;
  float x0 = (float)p[2 * i], x1 = (float)p[2 * i + 1];
  float ss = x0 * x0 + x1 * x1;
#pragma unroll
  for (int m = 16; m >= 1; m >>= 1) ss += __shfl_xor(ss, m, 32);
  const float r = rsqrtf(ss / (float)DH_ + EPSF);
  const float* wr_ = w + (size_t)br * wStride;
  x0 = x0 * r * wr_[2 * i];
  x1 = x1 * r * wr_[2 * i + 1];
  const float inv = __powf(10000.f, -(float)i / 32.f);
  const float ang = (float)n * inv;
  const float c = __cosf(ang), s = __sinf(ang);
  bf16* o = out + (((size_t)(br * B_ + b) * H_ + h) * NS + n) * DH_;
  o[2 * i]     = f2bf(x0 * c - x1 * s);
  o[2 * i + 1] = f2bf(x1 * c + x0 * s);
}

// ---------------------------------------------------------------------------
// 6) V transpose: strided bf16 in -> VT[(br*B+b)][h][d][n] bf16
// ---------------------------------------------------------------------------
__global__ __launch_bounds__(256) void vtrans_kernel(
    const bf16* __restrict__ in, bf16* __restrict__ out, long long sBr,
    long long sB, long long sH, long long sN, long long total) {
  for (long long t = (long long)blockIdx.x * blockDim.x + threadIdx.x;
       t < total; t += (long long)gridDim.x * blockDim.x) {
    const int d = (int)(t & 63);
    const int n = (int)((t >> 6) % NS);
    long long r = t / ((long long)64 * NS);
    const int h = (int)(r % H_);
    r /= H_;
    const int b = (int)(r % B_);
    const int br = (int)(r / B_);
    out[(((size_t)(br * B_ + b) * H_ + h) * DH_ + d) * NS + n] =
        in[(size_t)br * sBr + (size_t)b * sB + (size_t)h * sH +
           (size_t)n * sN + d];
  }
}

// ---------------------------------------------------------------------------
// 7) Causal flash attention, one wave per 32-query tile (2 M-subtiles so
//    every K/V fragment load feeds two WMMAs).
//    Q,K: [head][n][64] bf16 (RoPE'd). VT: [head][d][n] bf16.
//    out_mode 0: out[(gy)][n][64];  out_mode 1: out[b][n][h*64+d].
// ---------------------------------------------------------------------------
__global__ __launch_bounds__(32) void attn_kernel(
    const bf16* __restrict__ Q, const bf16* __restrict__ K,
    const bf16* __restrict__ VT, bf16* __restrict__ out, int BH,
    int out_mode) {
  __shared__ __align__(16) bf16 pbuf[2][16 * 32];
  const int lane = threadIdx.x;
  const int half = lane >> 4;
  const int lr   = lane & 15;
  const int qt = blockIdx.x;           // 32-row query tile
  const int gy = blockIdx.y;
  const int head = gy % BH;

  const bf16* Qh = Q  + (size_t)head * NS * DH_;
  const bf16* Kh = K  + (size_t)gy   * NS * DH_;
  const bf16* Vh = VT + (size_t)gy   * DH_ * NS;

  // Q fragments: 2 M-subtiles x 2 d-chunks of 16x32 A-fragments
  v16bf qf[2][2];
#pragma unroll
  for (int ms = 0; ms < 2; ++ms) {
    const bf16* qp = Qh + (size_t)(qt * 32 + ms * 16 + lr) * DH_;
#pragma unroll
    for (int c = 0; c < 2; ++c) {
      v8bf lo = *(const v8bf*)(qp + 32 * c + 16 * half);
      v8bf hi = *(const v8bf*)(qp + 32 * c + 8 + 16 * half);
      qf[ms][c] = cat8(lo, hi);
    }
  }

  v8f o[2][4];
  float mrow[2][8], lrow[2][8];
#pragma unroll
  for (int ms = 0; ms < 2; ++ms) {
#pragma unroll
    for (int ds = 0; ds < 4; ++ds) o[ms][ds] = vzero8();
#pragma unroll
    for (int r = 0; r < 8; ++r) {
      mrow[ms][r] = NEG_BIG;
      lrow[ms][r] = 0.f;
    }
  }

  const int kv_end = qt * 32 + 32;
  for (int kv0 = 0; kv0 < kv_end; kv0 += 32) {
    if (kv0 + 32 < kv_end) {
      __builtin_prefetch(Kh + (size_t)(kv0 + 32 + lane) * DH_, 0, 1);
      __builtin_prefetch(Vh + (size_t)lane * NS + kv0 + 32, 0, 1);
      __builtin_prefetch(Vh + (size_t)(32 + lane) * NS + kv0 + 32, 0, 1);
    }
    // K fragments for 32 keys, loaded once, used by both M-subtiles
    v16bf kf[2][2];
#pragma unroll
    for (int t = 0; t < 2; ++t)
#pragma unroll
      for (int c = 0; c < 2; ++c)
        kf[t][c] = *(const v16bf*)(Kh + (size_t)(kv0 + 16 * t + lr) * DH_ +
                                   32 * c + 16 * half);

    __syncthreads();  // pbuf free before restaging
#pragma unroll
    for (int ms = 0; ms < 2; ++ms) {
      // S = Q * K^T (two 16-key column tiles)
      v8f s[2] = {vzero8(), vzero8()};
#pragma unroll
      for (int t = 0; t < 2; ++t)
#pragma unroll
        for (int c = 0; c < 2; ++c)
          s[t] = __builtin_amdgcn_wmma_f32_16x16x32_bf16(
              false, qf[ms][c], false, kf[t][c], (short)0, s[t], false, false);

      // scale + causal mask
#pragma unroll
      for (int t = 0; t < 2; ++t)
#pragma unroll
        for (int r = 0; r < 8; ++r) {
          const int key  = kv0 + 16 * t + lr;
          const int qrow = qt * 32 + ms * 16 + r + 8 * half;
          const float v = s[t][r] * 0.125f;  // 1/sqrt(64)
          s[t][r] = (key <= qrow) ? v : NEG_BIG;
        }

      // online softmax (rows live across the 16-lane half-groups)
      float alpha[8];
#pragma unroll
      for (int r = 0; r < 8; ++r) {
        float mx = fmaxf(s[0][r], s[1][r]);
#pragma unroll
        for (int mk = 1; mk < 16; mk <<= 1)
          mx = fmaxf(mx, __shfl_xor(mx, mk, 32));
        const float mnew = fmaxf(mrow[ms][r], mx);
        alpha[r] = __expf(mrow[ms][r] - mnew);
        mrow[ms][r] = mnew;
        const float p0 = __expf(s[0][r] - mnew);
        const float p1 = __expf(s[1][r] - mnew);
        s[0][r] = p0;
        s[1][r] = p1;
        float rs = p0 + p1;
#pragma unroll
        for (int mk = 1; mk < 16; mk <<= 1) rs += __shfl_xor(rs, mk, 32);
        lrow[ms][r] = lrow[ms][r] * alpha[r] + rs;
      }

#pragma unroll
      for (int ds = 0; ds < 4; ++ds)
#pragma unroll
        for (int r = 0; r < 8; ++r) o[ms][ds][r] *= alpha[r];

      // stage P (C-layout -> row-major LDS)
#pragma unroll
      for (int t = 0; t < 2; ++t)
#pragma unroll
        for (int r = 0; r < 8; ++r)
          pbuf[ms][(r + 8 * half) * 32 + 16 * t + lr] = f2bf(s[t][r]);
    }
    __syncthreads();

    // reload P in A-fragment layout
    v16bf pf[2];
#pragma unroll
    for (int ms = 0; ms < 2; ++ms) {
      const bf16* pp = pbuf[ms] + lr * 32;
      v8bf lo = *(const v8bf*)(pp + 16 * half);
      v8bf hi = *(const v8bf*)(pp + 8 + 16 * half);
      pf[ms] = cat8(lo, hi);
    }

    // O += P * V: each V fragment load feeds both M-subtiles
#pragma unroll
    for (int ds = 0; ds < 4; ++ds) {
      const v16bf vf =
          *(const v16bf*)(Vh + (size_t)(ds * 16 + lr) * NS + kv0 + 16 * half);
#pragma unroll
      for (int ms = 0; ms < 2; ++ms)
        o[ms][ds] = __builtin_amdgcn_wmma_f32_16x16x32_bf16(
            false, pf[ms], false, vf, (short)0, o[ms][ds], false, false);
    }
  }

  size_t obase;
  int ostride;
  if (out_mode == 0) {
    obase = (size_t)gy * NS * DH_;
    ostride = DH_;
  } else {
    const int b = head / H_, h = head % H_;
    obase = (size_t)b * NS * (H_ * DH_) + (size_t)h * DH_;
    ostride = H_ * DH_;
  }
#pragma unroll
  for (int ms = 0; ms < 2; ++ms)
#pragma unroll
    for (int r = 0; r < 8; ++r) {
      const float inv = 1.f / lrow[ms][r];
      const int qrow = qt * 32 + ms * 16 + r + 8 * half;
#pragma unroll
      for (int ds = 0; ds < 4; ++ds)
        out[obase + (size_t)qrow * ostride + ds * 16 + lr] =
            f2bf(o[ms][ds][r] * inv);
    }
}

// ---------------------------------------------------------------------------
// Host orchestration
// ---------------------------------------------------------------------------
extern "C" void kernel_launch(void* const* d_in, const int* in_sizes, int n_in,
                              void* d_out, int out_size, void* d_ws,
                              size_t ws_size, hipStream_t stream) {
  (void)in_sizes; (void)n_in; (void)out_size; (void)ws_size;
  const float* tokens  = (const float*)d_in[0];
  const float* w_norm  = (const float*)d_in[1];
  const float* w_q     = (const float*)d_in[2];
  const float* w_k     = (const float*)d_in[3];
  const float* w_v     = (const float*)d_in[4];
  const float* w_keyn  = (const float*)d_in[5];
  const float* w_nkn   = (const float*)d_in[6];
  const float* w_out   = (const float*)d_in[7];
  float* outp = (float*)d_out;

  char* ws = (char*)d_ws;
  const size_t MB = 1ull << 20;
  bf16* xb  = (bf16*)(ws + 0 * MB);     // [4096][1024]       8 MB
  bf16* wqT = (bf16*)(ws + 8 * MB);     // [1024][1024]       2 MB
  bf16* wkT = (bf16*)(ws + 10 * MB);    // [3072][1024]       6 MB
  bf16* wvT = (bf16*)(ws + 16 * MB);    // [3072][1024]       6 MB
  bf16* woT = (bf16*)(ws + 22 * MB);    // [1024][1024]       2 MB
  bf16* qg  = (bf16*)(ws + 24 * MB);    // [4096][1024]       8 MB
  bf16* kg  = (bf16*)(ws + 32 * MB);    // [4096][3072]      24 MB
  bf16* vg  = (bf16*)(ws + 56 * MB);    // [4096][3072]      24 MB
  bf16* QB  = (bf16*)(ws + 80 * MB);    // [32][2048][64]     8 MB
  bf16* KB  = (bf16*)(ws + 88 * MB);    // [96][2048][64]    24 MB
  bf16* VTB = (bf16*)(ws + 112 * MB);   // [96][64][2048]    24 MB
  bf16* NO  = (bf16*)(ws + 24 * MB);    // [3][32][2048][64] 24 MB (reuse qkv)
  bf16* Q2  = (bf16*)(ws + 48 * MB);    // [32][2048][64]     8 MB
  bf16* K2  = (bf16*)(ws + 56 * MB);    // [32][2048][64]     8 MB (reuse vg)
  bf16* VT2 = (bf16*)(ws + 64 * MB);    // [32][64][2048]     8 MB
  bf16* O2  = (bf16*)(ws + 72 * MB);    // [4096][1024]       8 MB

  // 1) x = rmsnorm(tokens) -> bf16
  rmsnorm_rows_kernel<<<MROWS, 256, 0, stream>>>(tokens, w_norm, xb, DIM_);

  // 2) transpose+cast weights
  transpose_cast_kernel<<<4096, 256, 0, stream>>>(w_q, wqT, DIM_, DIM_);
  transpose_cast_kernel<<<4096, 256, 0, stream>>>(w_k, wkT, DIM_, 3 * DIM_);
  transpose_cast_kernel<<<4096, 256, 0, stream>>>(w_v, wvT, DIM_, 3 * DIM_);
  transpose_cast_kernel<<<4096, 256, 0, stream>>>(w_out, woT, DIM_, DIM_);

  // 3) QKV GEMMs (bf16 WMMA, bf16 outputs)
  gemm_bf16_kernel<bf16><<<dim3(DIM_ / 256, MROWS / 64), 256, 0, stream>>>(
      xb, wqT, qg, MROWS, DIM_, DIM_);
  gemm_bf16_kernel<bf16><<<dim3(3 * DIM_ / 256, MROWS / 64), 256, 0, stream>>>(
      xb, wkT, kg, MROWS, DIM_, 3 * DIM_);
  gemm_bf16_kernel<bf16><<<dim3(3 * DIM_ / 256, MROWS / 64), 256, 0, stream>>>(
      xb, wvT, vg, MROWS, DIM_, 3 * DIM_);

  // 4) Stage-1 pre-attention: rope(q), rmsnorm+rope(k), transpose(v)
  // q layout: [(b*NS+n)][h*64+d]
  rope_cast_kernel<<<B_ * H_ * NS, 32, 0, stream>>>(
      qg, QB, (long long)NS * 1024, 64, 1024);
  // k/v layout: [(b*NS+n)][br*1024 + h*64 + d]
  knorm_rope_kernel<<<3 * B_ * H_ * NS, 32, 0, stream>>>(
      kg, w_keyn, KB, 1024, (long long)NS * 3072, 64, 3072, DH_);
  vtrans_kernel<<<8192, 256, 0, stream>>>(
      vg, VTB, 1024, (long long)NS * 3072, 64, 3072,
      (long long)3 * B_ * H_ * NS * DH_);

  // 5) Stage-1 attention: 3 branches share Q
  attn_kernel<<<dim3(NS / 32, 3 * B_ * H_), 32, 0, stream>>>(
      QB, KB, VTB, NO, B_ * H_, 0);

  // 6) Stage-2 pre-attention from NO[br][b][h][n][d]
  const long long BHND = (long long)B_ * H_ * NS * DH_;
  rope_cast_kernel<<<B_ * H_ * NS, 32, 0, stream>>>(
      NO + 0 * BHND, Q2, (long long)H_ * NS * DH_, (long long)NS * DH_, DH_);
  knorm_rope_kernel<<<B_ * H_ * NS, 32, 0, stream>>>(
      NO + 1 * BHND, w_nkn, K2, 0, (long long)H_ * NS * DH_,
      (long long)NS * DH_, DH_, 0);
  vtrans_kernel<<<8192, 256, 0, stream>>>(
      NO + 2 * BHND, VT2, 0, (long long)H_ * NS * DH_, (long long)NS * DH_,
      DH_, BHND);

  // 7) Stage-2 attention, output directly in [b][n][h*64+d] layout
  attn_kernel<<<dim3(NS / 32, B_ * H_), 32, 0, stream>>>(
      Q2, K2, VT2, O2, B_ * H_, 1);

  // 8) Final GEMM: out = O2 @ w_out  (fp32 output)
  gemm_bf16_kernel<float><<<dim3(DIM_ / 256, MROWS / 64), 256, 0, stream>>>(
      O2, woT, outp, MROWS, DIM_, DIM_);
}
